// Decoder_32169305047767
// MI455X (gfx1250) — matrix operand order repaired and checked by reading, not compiled, not run on hardware
//
#include <hip/hip_runtime.h>

typedef _Float16 h16;
typedef __attribute__((ext_vector_type(16))) _Float16 v16h;
typedef __attribute__((ext_vector_type(8)))  float    v8f;

#define NB   32      // batch
#define NS   512     // encoder steps
#define NT   800     // decoder steps
#define NDE  512     // encoder dim
#define NH   1024    // LSTM hidden
#define NPRE 256
#define NMEL 80
#define SPLITG 8     // split-K slices for LSTM gate GEMMs
#define SPLITE 4     // split-K slices for attention-score GEMM

static __device__ __forceinline__ float sigm(float x) { return 1.0f / (1.0f + __expf(-x)); }

// ---------------------------------------------------------------------------
// WMMA f16 GEMM:  Y[M,N] = act( sum_tap A_shift[M,K] * W[N, tap, K]^T + bias )
// A row-major f16 (lda halves), W row-major f16 (ldw halves, taps at +tap*K).
// Block: 256 thr = 8 waves; block tile 32M x 256N; wave tile 16M x 64N.
// gridDim.z = split-K slices (SM==0 only); slice s stores at out0 + s*M*ld0.
// Out-of-range N-tiles / halo rows read a zeroed scratch buffer instead of
// branching, so the K-loop is branch-free. All 5 fragment-load pairs of a
// K-step are issued before the 4 WMMAs so waits can be partial/overlapped.
// SM: 0 f32 row-major(ld0) (+weight prefetch-ahead); 1 f16 row-major(ld0);
//     2 mel: out0 f32 [B,80,T] (m = t*32+b) + out1 f16 im2row [B*T,96];
//     3 f32 channel-major out0[n*M + m]
// ---------------------------------------------------------------------------
template <int NTAPS, int ACT, int SM, bool NB_CHECK>
__global__ __launch_bounds__(256) void gemm_wmma(
    const h16* __restrict__ A, int lda,
    const h16* __restrict__ W, int ldw,
    const float* __restrict__ bias,
    int M, int N, int K, int rowPeriod,
    void* __restrict__ out0, void* __restrict__ out1, int ld0,
    const h16* __restrict__ zbuf)
{
  const int lane  = threadIdx.x & 31;
  const int wave  = threadIdx.x >> 5;
  const int lm    = lane & 15;
  const int lg    = lane >> 4;
  const int m0    = blockIdx.y * 32 + (wave >> 2) * 16;
  const int nBase = blockIdx.x * 256 + (wave & 3) * 64;
  const int kSlice = K / gridDim.z;
  const int kBeg  = blockIdx.z * kSlice;
  const int kEnd  = kBeg + kSlice;
  const int mr    = m0 + lm;

  // B row base pointers; N-tile validity is uniform per 16-lane group
  const h16* bRow[4];
  bool tileOK[4];
#pragma unroll
  for (int j = 0; j < 4; ++j) {
    const int n = nBase + j * 16 + lm;
    tileOK[j] = (!NB_CHECK) || ((nBase + j * 16) < N);
    bRow[j] = tileOK[j] ? (W + (long)n * ldw) : zbuf;
  }

  v8f acc[4] = {};
  union Frag { uint4 u[2]; v16h v; };

#pragma unroll
  for (int tap = 0; tap < NTAPS; ++tap) {
    const int shift = tap - NTAPS / 2;
    const h16* aBase = A + (long)(mr + shift) * lda;
    if (NTAPS > 1) {
      int tloc = (mr % rowPeriod) + shift;
      if (tloc < 0 || tloc >= rowPeriod) aBase = zbuf;  // cndmask on address
    }
    const long tapOff = (long)tap * K;
    for (int k0 = kBeg; k0 < kEnd; k0 += 32) {
      Frag af;
      const uint4* ap = (const uint4*)(aBase + k0 + lg * 16);
      af.u[0] = ap[0];            // halves 0..7  -> K = k0 + lg*16 + 0..7
      af.u[1] = ap[1];            // halves 8..15 -> K = k0 + lg*16 + 8..15
      Frag bf[4];
#pragma unroll
      for (int j = 0; j < 4; ++j) {
        const uint4* bp = (const uint4*)(bRow[j] + tapOff + k0 + lg * 16);
        bf[j].u[0] = bp[0];       // halves 0..15 -> K = k0 + lg*16 + 0..15
        bf[j].u[1] = bp[1];
      }
      if constexpr (SM == 0) {    // latency-critical sequential GEMMs:
#pragma unroll                    // prefetch next K-chunk of the weight rows
        for (int j = 0; j < 4; ++j)
          __builtin_prefetch(bRow[j] + tapOff + k0 + 64, 0, 0);
      }
#pragma unroll
      for (int j = 0; j < 4; ++j)
        acc[j] = __builtin_amdgcn_wmma_f32_16x16x32_f16(
            false, af.v, false, bf[j].v, (short)0, acc[j], false, false);
    }
  }

#pragma unroll
  for (int j = 0; j < 4; ++j) {
    if (NB_CHECK && !tileOK[j]) continue;   // wave-uniform
    const int n = nBase + j * 16 + lm;
    const float bv = bias ? bias[n] : 0.0f;
#pragma unroll
    for (int r = 0; r < 8; ++r) {
      const int m = m0 + lg * 8 + r;        // D layout: m = VGPR + (lane/16)*8
      float y = acc[j][r] + bv;
      if (ACT == 1) y = fmaxf(y, 0.0f);
      if constexpr (SM == 0) {
        ((float*)out0)[(long)blockIdx.z * M * ld0 + (long)m * ld0 + n] = y;
      } else if constexpr (SM == 1) {
        ((h16*)out0)[(long)m * ld0 + n] = (h16)y;
      } else if constexpr (SM == 2) {
        const int b = m & 31, t = m >> 5;
        ((float*)out0)[((long)b * NMEL + n) * NT + t] = y;
        ((h16*)out1)[((long)b * NT + t) * 96 + n] = (h16)y;
      } else {
        ((float*)out0)[(long)n * M + m] = y;
      }
    }
  }
}

// ------------------------- small utility kernels ---------------------------
__global__ void k_cvt(const float* __restrict__ s, h16* __restrict__ d, int n) {
  int i = blockIdx.x * 256 + threadIdx.x; if (i < n) d[i] = (h16)s[i];
}
__global__ void k_pack2(const float* __restrict__ A, const float* __restrict__ Bp,
                        h16* __restrict__ d, int N, int K1, int K2) {
  int i = blockIdx.x * 256 + threadIdx.x; int KT = K1 + K2;
  if (i >= N * KT) return;
  int n = i / KT, k = i % KT;
  float v = (k < K1) ? A[(long)n * K1 + k] : Bp[(long)n * K2 + (k - K1)];
  d[i] = (h16)v;
}
__global__ void k_padcvt(const float* __restrict__ s, h16* __restrict__ d, int N, int Kin, int Kp) {
  int i = blockIdx.x * 256 + threadIdx.x; if (i >= N * Kp) return;
  int n = i / Kp, k = i % Kp;
  d[i] = (k < Kin) ? (h16)s[(long)n * Kin + k] : (h16)0.0f;
}
__global__ void k_convw(const float* __restrict__ s, h16* __restrict__ d, int CO, int CI, int CIP) {
  int i = blockIdx.x * 256 + threadIdx.x; int per = 5 * CIP;
  if (i >= CO * per) return;
  int c = i / per, r = i % per, tap = r / CIP, k = r % CIP;
  d[i] = (k < CI) ? (h16)s[((long)c * CI + k) * 5 + tap] : (h16)0.0f;
}
__global__ void k_add2(const float* a, const float* b, float* d, int n) {
  int i = blockIdx.x * 256 + threadIdx.x; if (i < n) d[i] = a[i] + b[i];
}
__global__ void k_zeroh(h16* d, int n) {
  int i = blockIdx.x * 256 + threadIdx.x; if (i < n) d[i] = (h16)0.0f;
}
__global__ void k_zerof(float* d, int n) {
  int i = blockIdx.x * 256 + threadIdx.x; if (i < n) d[i] = 0.0f;
}
__global__ void k_mask(const float* __restrict__ enc, float* __restrict__ m) {
  int i = blockIdx.x * 256 + threadIdx.x; if (i >= NB * NS) return;
  const float* p = enc + (long)i * NDE;
  float s = 0.0f;
  for (int d2 = 0; d2 < NDE; ++d2) s += p[d2];
  m[i] = (s == 0.0f) ? 1.0f : 0.0f;
}
__global__ void k_stage_melin(const float* __restrict__ melT, h16* __restrict__ d) {
  int i = blockIdx.x * 256 + threadIdx.x; if (i >= NT * NB * 96) return;
  int m = i / 96, c = i % 96;
  int t = m >> 5, b = m & 31;
  float v = (c < NMEL && t > 0) ? melT[((long)b * NMEL + c) * NT + (t - 1)] : 0.0f;
  d[i] = (h16)v;
}
__global__ void k_stage(const h16* __restrict__ pre, h16* __restrict__ xa,
                        h16* __restrict__ x0c, int t) {
  int i = blockIdx.x * 256 + threadIdx.x; if (i >= NB * NPRE) return;
  int b = i >> 8, c = i & 255;
  h16 v = pre[((long)t * NB + b) * NPRE + c];
  xa[b * 1280 + c]  = v;
  x0c[b * 1792 + c] = v;
}

// attention: sum split-K partials + bias, mask, softmax, context.
// One block per batch row; 512 threads (thread = s, then thread = d).
__global__ __launch_bounds__(512) void k_att(
    const float* __restrict__ e, const float* __restrict__ ba,
    const float* __restrict__ mask, const float* __restrict__ enc,
    float* __restrict__ alignOut, h16* __restrict__ x0c, h16* __restrict__ feat, int t)
{
  int b = blockIdx.x, tid = threadIdx.x;
  __shared__ float sm[NS];
  __shared__ float red[16];
  int wv = tid >> 5, ln = tid & 31;

  float ev = ba[tid];
#pragma unroll
  for (int s = 0; s < SPLITE; ++s) ev += e[(long)s * NB * NS + b * NS + tid];
  if (mask[b * NS + tid] != 0.0f) ev = -__builtin_inff();
  float mx = ev;
  for (int o = 16; o; o >>= 1) mx = fmaxf(mx, __shfl_xor(mx, o, 32));
  if (ln == 0) red[wv] = mx;
  __syncthreads();
  if (tid == 0) { float m2 = red[0]; for (int i = 1; i < 16; ++i) m2 = fmaxf(m2, red[i]); red[0] = m2; }
  __syncthreads();
  float MX = red[0];
  __syncthreads();
  float p = __expf(ev - MX);
  float su = p;
  for (int o = 16; o; o >>= 1) su += __shfl_xor(su, o, 32);
  if (ln == 0) red[wv] = su;
  __syncthreads();
  if (tid == 0) { float S = 0.0f; for (int i = 0; i < 16; ++i) S += red[i]; red[0] = S; }
  __syncthreads();
  float aw = p / red[0];
  sm[tid] = aw;
  alignOut[((long)b * NS + tid) * NT + t] = aw;
  __syncthreads();
  const float* eb = enc + (long)b * NS * NDE + tid;   // ctx[b,d]; d = tid
  float s = 0.0f;
  for (int si = 0; si < NS; ++si) s += sm[si] * eb[(long)si * NDE];
  x0c[b * 1792 + NPRE + tid] = (h16)s;
  feat[((long)t * NB + b) * 1536 + NH + tid] = (h16)s;
}

// LSTM pointwise: sum split-K gate partials + combined bias, update c/h.
__global__ void k_pw0(const float* __restrict__ g, const float* __restrict__ bb,
                      float* __restrict__ c0, h16* __restrict__ xa,
                      h16* __restrict__ x0c, h16* __restrict__ x1c) {
  int i = blockIdx.x * 256 + threadIdx.x; if (i >= NB * NH) return;
  int b = i >> 10, j = i & 1023;
  const float* gb = g + (long)b * 4096;
  const int SL = NB * 4096;
  float vi = bb[j], vf = bb[1024 + j], vg = bb[2048 + j], vo = bb[3072 + j];
#pragma unroll
  for (int s = 0; s < SPLITG; ++s) {
    const float* p = gb + (long)s * SL;
    vi += p[j]; vf += p[1024 + j]; vg += p[2048 + j]; vo += p[3072 + j];
  }
  float cn = sigm(vf) * c0[i] + sigm(vi) * tanhf(vg);
  float hn = sigm(vo) * tanhf(cn);
  c0[i] = cn;
  h16 h = (h16)hn;
  xa[b * 1280 + NPRE + j] = h;
  x0c[b * 1792 + 768 + j] = h;
  x1c[b * 2048 + j] = h;
}
__global__ void k_pw1(const float* __restrict__ g, const float* __restrict__ bb,
                      float* __restrict__ c1, h16* __restrict__ x1c,
                      h16* __restrict__ feat, int t) {
  int i = blockIdx.x * 256 + threadIdx.x; if (i >= NB * NH) return;
  int b = i >> 10, j = i & 1023;
  const float* gb = g + (long)b * 4096;
  const int SL = NB * 4096;
  float vi = bb[j], vf = bb[1024 + j], vg = bb[2048 + j], vo = bb[3072 + j];
#pragma unroll
  for (int s = 0; s < SPLITG; ++s) {
    const float* p = gb + (long)s * SL;
    vi += p[j]; vf += p[1024 + j]; vg += p[2048 + j]; vo += p[3072 + j];
  }
  float cn = sigm(vf) * c1[i] + sigm(vi) * tanhf(vg);
  float hn = sigm(vo) * tanhf(cn);
  c1[i] = cn;
  h16 h = (h16)hn;
  x1c[b * 2048 + 1024 + j] = h;
  feat[((long)t * NB + b) * 1536 + j] = h;
}

__global__ void k_stop(const h16* __restrict__ feat, const h16* __restrict__ ws,
                       const float* __restrict__ bs, float* __restrict__ out) {
  int i = blockIdx.x * 256 + threadIdx.x; if (i >= NT * NB) return;
  int t = i >> 5, b = i & 31;
  const h16* f = feat + (long)i * 1536;
  float s = bs[0];
  for (int k = 0; k < 1536; ++k) s += (float)f[k] * (float)ws[k];
  out[b * NT + t] = 1.0f / (1.0f + __expf(-s));
}

__global__ __launch_bounds__(256) void k_bnstats(const float* __restrict__ y,
                                                 float* __restrict__ stats, int Mtot) {
  int c = blockIdx.x;
  const float* p = y + (long)c * Mtot;
  float s = 0.0f, s2 = 0.0f;
  for (int i = threadIdx.x; i < Mtot; i += 256) { float v = p[i]; s += v; s2 += v * v; }
  for (int o = 16; o; o >>= 1) { s += __shfl_xor(s, o, 32); s2 += __shfl_xor(s2, o, 32); }
  __shared__ float ls[8], ls2[8];
  int wv = threadIdx.x >> 5;
  if ((threadIdx.x & 31) == 0) { ls[wv] = s; ls2[wv] = s2; }
  __syncthreads();
  if (threadIdx.x == 0) {
    float S = 0.0f, S2 = 0.0f;
    for (int i = 0; i < 8; ++i) { S += ls[i]; S2 += ls2[i]; }
    float mean = S / Mtot;
    stats[2 * c] = mean;
    stats[2 * c + 1] = S2 / Mtot - mean * mean;
  }
}

__global__ void k_bnapply(const float* __restrict__ y, const float* __restrict__ stats,
                          const float* __restrict__ g, const float* __restrict__ be,
                          h16* __restrict__ nextIn, int ldn,
                          float* __restrict__ outAdd, const float* __restrict__ melBase,
                          int C, int Mtot, int act) {
  long i = (long)blockIdx.x * 256 + threadIdx.x;
  if (i >= (long)C * Mtot) return;
  int c = (int)(i / Mtot), m = (int)(i % Mtot);
  float mean = stats[2 * c], var = stats[2 * c + 1];
  float v = (y[i] - mean) * rsqrtf(var + 1e-5f) * g[c] + be[c];
  if (act) v = tanhf(v);
  if (nextIn) nextIn[(long)m * ldn + c] = (h16)v;
  if (outAdd) {
    int b = m / NT, t = m % NT;
    long o = ((long)b * NMEL + c) * NT + t;
    outAdd[o] = melBase[o] + v;
  }
}

// ---------------------------------------------------------------------------
extern "C" void kernel_launch(void* const* d_in, const int* in_sizes, int n_in,
                              void* d_out, int out_size, void* d_ws, size_t ws_size,
                              hipStream_t stream)
{
  (void)in_sizes; (void)n_in; (void)out_size; (void)ws_size;
  const float* enc  = (const float*)d_in[0];
  const float* melT = (const float*)d_in[1];
  const float* Wa   = (const float*)d_in[2];
  const float* ba   = (const float*)d_in[3];
  const float* Wp1  = (const float*)d_in[4];
  const float* bp1  = (const float*)d_in[5];
  const float* Wp2  = (const float*)d_in[6];
  const float* bp2  = (const float*)d_in[7];
  const float* Wih0 = (const float*)d_in[8];
  const float* Whh0 = (const float*)d_in[9];
  const float* bih0 = (const float*)d_in[10];
  const float* bhh0 = (const float*)d_in[11];
  const float* Wih1 = (const float*)d_in[12];
  const float* Whh1 = (const float*)d_in[13];
  const float* bih1 = (const float*)d_in[14];
  const float* bhh1 = (const float*)d_in[15];
  const float* Wt   = (const float*)d_in[16];
  const float* bt   = (const float*)d_in[17];
  const float* Wsw  = (const float*)d_in[18];
  const float* bs   = (const float*)d_in[19];
  const float *cw[5], *cb[5], *gg[5], *bev[5];
  for (int i = 0; i < 5; ++i) {
    cw[i]  = (const float*)d_in[20 + 4 * i];
    cb[i]  = (const float*)d_in[21 + 4 * i];
    gg[i]  = (const float*)d_in[22 + 4 * i];
    bev[i] = (const float*)d_in[23 + 4 * i];
  }

  size_t off = 0;
  auto alloc  = [&](size_t bytes) -> void* {
    off = (off + 255) & ~(size_t)255;
    void* p = (char*)d_ws + off;
    off += bytes;
    return p;
  };
  auto allocH = [&](size_t n) { return (h16*)alloc(n * sizeof(h16)); };
  auto allocF = [&](size_t n) { return (float*)alloc(n * sizeof(float)); };
  auto cdiv   = [](int a, int b) { return (a + b - 1) / b; };

  const int MR = NT * NB;  // 25600 rows for time-parallel GEMMs

  h16* W0c  = allocH((size_t)4096 * 1792);
  h16* W1c  = allocH((size_t)4096 * 2048);
  h16* Wa16 = allocH((size_t)NS * 1280);
  h16* Wp1p = allocH(256 * 96);
  h16* Wp2h = allocH(256 * 256);
  h16* Wt16 = allocH(NMEL * 1536);
  h16* Ws16 = allocH(1536);
  int cin[5]  = {80, 512, 512, 512, 512};
  int cinp[5] = {96, 512, 512, 512, 512};
  int cou[5]  = {512, 512, 512, 512, 80};
  h16* Wc[5];
  for (int i = 0; i < 5; ++i) Wc[i] = allocH((size_t)cou[i] * 5 * cinp[i]);
  h16* melin = allocH((size_t)MR * 96);
  h16* pre1  = allocH((size_t)MR * 256);
  h16* pre   = allocH((size_t)MR * 256);
  h16* feat  = allocH((size_t)MR * 1536);
  h16* xa    = allocH(NB * 1280);
  h16* x0c   = allocH(NB * 1792);
  h16* x1c   = allocH(NB * 2048);
  h16* xt0   = allocH((size_t)MR * 96);
  h16* xt    = allocH((size_t)MR * 512);
  h16* zbuf  = allocH(4096);                 // 8 KB zero scratch for OOB frags
  float* b01  = allocF(4096);
  float* b11  = allocF(4096);
  float* ebuf = allocF((size_t)SPLITE * NB * NS);    // split-K partials
  float* g0b  = allocF((size_t)SPLITG * NB * 4096);
  float* g1b  = allocF((size_t)SPLITG * NB * 4096);
  float* c0   = allocF(NB * NH);
  float* c1   = allocF(NB * NH);
  float* maskb = allocF(NB * NS);
  float* ybuf  = allocF((size_t)512 * MR);
  float* stats = allocF(1024);

  float* out_mel   = (float*)d_out;
  float* out_post  = out_mel + (size_t)NB * NMEL * NT;
  float* out_stop  = out_post + (size_t)NB * NMEL * NT;
  float* out_align = out_stop + (size_t)NB * NT;

  // --- weight conversion / packing (f32 -> f16, K padded to /32) ---
  k_pack2<<<cdiv(4096 * 1792, 256), 256, 0, stream>>>(Wih0, Whh0, W0c, 4096, 768, 1024);
  k_pack2<<<cdiv(4096 * 2048, 256), 256, 0, stream>>>(Wih1, Whh1, W1c, 4096, 1024, 1024);
  k_cvt<<<cdiv(NS * 1280, 256), 256, 0, stream>>>(Wa, Wa16, NS * 1280);
  k_padcvt<<<cdiv(256 * 96, 256), 256, 0, stream>>>(Wp1, Wp1p, 256, 80, 96);
  k_cvt<<<cdiv(256 * 256, 256), 256, 0, stream>>>(Wp2, Wp2h, 256 * 256);
  k_cvt<<<cdiv(NMEL * 1536, 256), 256, 0, stream>>>(Wt, Wt16, NMEL * 1536);
  k_cvt<<<cdiv(1536, 256), 256, 0, stream>>>(Wsw, Ws16, 1536);
  for (int i = 0; i < 5; ++i)
    k_convw<<<cdiv(cou[i] * 5 * cinp[i], 256), 256, 0, stream>>>(cw[i], Wc[i], cou[i], cin[i], cinp[i]);
  k_add2<<<cdiv(4096, 256), 256, 0, stream>>>(bih0, bhh0, b01, 4096);
  k_add2<<<cdiv(4096, 256), 256, 0, stream>>>(bih1, bhh1, b11, 4096);

  // --- state init (deterministic per call) ---
  k_zeroh<<<cdiv(NB * 1280, 256), 256, 0, stream>>>(xa, NB * 1280);
  k_zeroh<<<cdiv(NB * 1792, 256), 256, 0, stream>>>(x0c, NB * 1792);
  k_zeroh<<<cdiv(NB * 2048, 256), 256, 0, stream>>>(x1c, NB * 2048);
  k_zeroh<<<cdiv(MR * 96, 256), 256, 0, stream>>>(xt0, MR * 96);
  k_zeroh<<<cdiv(4096, 256), 256, 0, stream>>>(zbuf, 4096);
  k_zerof<<<cdiv(NB * NH, 256), 256, 0, stream>>>(c0, NB * NH);
  k_zerof<<<cdiv(NB * NH, 256), 256, 0, stream>>>(c1, NB * NH);
  k_mask<<<cdiv(NB * NS, 256), 256, 0, stream>>>(enc, maskb);

  // --- prenet (time-parallel, WMMA, f16 out) ---
  k_stage_melin<<<cdiv(MR * 96, 256), 256, 0, stream>>>(melT, melin);
  gemm_wmma<1, 1, 1, false><<<dim3(1, MR / 32, 1), 256, 0, stream>>>(
      melin, 96, Wp1p, 96, bp1, MR, 256, 96, 0, pre1, nullptr, 256, zbuf);
  gemm_wmma<1, 1, 1, false><<<dim3(1, MR / 32, 1), 256, 0, stream>>>(
      pre1, 256, Wp2h, 256, bp2, MR, 256, 256, 0, pre, nullptr, 256, zbuf);

  // --- sequential decoder scan (split-K for parallelism on the chain) ---
  for (int t = 0; t < NT; ++t) {
    k_stage<<<NB, 256, 0, stream>>>(pre, xa, x0c, t);
    gemm_wmma<1, 0, 0, false><<<dim3(2, 1, SPLITE), 256, 0, stream>>>(
        xa, 1280, Wa16, 1280, nullptr, NB, NS, 1280, 0, ebuf, nullptr, NS, zbuf);
    k_att<<<NB, 512, 0, stream>>>(ebuf, ba, maskb, enc, out_align, x0c, feat, t);
    gemm_wmma<1, 0, 0, false><<<dim3(16, 1, SPLITG), 256, 0, stream>>>(
        x0c, 1792, W0c, 1792, nullptr, NB, 4096, 1792, 0, g0b, nullptr, 4096, zbuf);
    k_pw0<<<NB * NH / 256, 256, 0, stream>>>(g0b, b01, c0, xa, x0c, x1c);
    gemm_wmma<1, 0, 0, false><<<dim3(16, 1, SPLITG), 256, 0, stream>>>(
        x1c, 2048, W1c, 2048, nullptr, NB, 4096, 2048, 0, g1b, nullptr, 4096, zbuf);
    k_pw1<<<NB * NH / 256, 256, 0, stream>>>(g1b, b11, c1, x1c, feat, t);
  }

  // --- output projections ---
  gemm_wmma<1, 0, 2, true><<<dim3(1, MR / 32, 1), 256, 0, stream>>>(
      feat, 1536, Wt16, 1536, bt, MR, NMEL, 1536, 0, out_mel, xt0, 0, zbuf);
  k_stop<<<cdiv(MR, 256), 256, 0, stream>>>(feat, Ws16, bs, out_stop);

  // --- postnet: 5-tap implicit-GEMM conv + training BN (+tanh on 0..3) ---
  for (int l = 0; l < 5; ++l) {
    const h16* Ain = (l == 0) ? xt0 : xt;
    int Kl = cinp[l];
    if (l < 4)
      gemm_wmma<5, 0, 3, false><<<dim3(2, MR / 32, 1), 256, 0, stream>>>(
          Ain, Kl, Wc[l], 5 * Kl, cb[l], MR, cou[l], Kl, NT, ybuf, nullptr, 0, zbuf);
    else
      gemm_wmma<5, 0, 3, true><<<dim3(1, MR / 32, 1), 256, 0, stream>>>(
          Ain, Kl, Wc[l], 5 * Kl, cb[l], MR, cou[l], Kl, NT, ybuf, nullptr, 0, zbuf);
    k_bnstats<<<cou[l], 256, 0, stream>>>(ybuf, stats, MR);
    if (l < 4)
      k_bnapply<<<cdiv(cou[l] * MR, 256), 256, 0, stream>>>(ybuf, stats, gg[l], bev[l],
          xt, 512, nullptr, nullptr, cou[l], MR, 1);
    else
      k_bnapply<<<cdiv(cou[l] * MR, 256), 256, 0, stream>>>(ybuf, stats, gg[l], bev[l],
          nullptr, 0, out_post, out_mel, cou[l], MR, 0);
  }
}
